// ViTBlock_79783312490676
// MI455X (gfx1250) — matrix operand order repaired and checked by reading, not compiled, not run on hardware
//
#include <hip/hip_runtime.h>

typedef _Float16 f16;
typedef _Float16 v8h  __attribute__((ext_vector_type(8)));
typedef _Float16 v16h __attribute__((ext_vector_type(16)));
typedef float    v8f  __attribute__((ext_vector_type(8)));
typedef float    v4f  __attribute__((ext_vector_type(4)));

#define DIM   96
#define TOK   64
#define LDX   104   // f16 stride for [64][96] buffers (208B rows, 16B aligned)
#define LDV   72    // f16 stride for vT [96][64] and attn [64][64] (144B rows)
#define LDSC  66    // f32 stride for scores
#define LDL   100   // f32 stride for lepe
#define LDO   68    // f32 stride for output transpose buffer (272B rows, 16B aligned)

// dynamic-LDS byte offsets (all 16B aligned)
#define OFF_XS   0        // [64][LDX] f16, reused as O-hat for out proj
#define OFF_Q    13312    // [64][LDX] f16 (pre-scaled by 1/sqrt(96))
#define OFF_K    26624    // [64][LDX] f16
#define OFF_VT   39936    // [96][LDV] f16 (V transposed: [ch][token])
#define OFF_ATT  53760    // [64][LDV] f16
#define OFF_SC   62976    // [64][LDSC] f32
#define OFF_U    79872    // union: x-stage [96][64] f32 (24576) | lep [64][LDL] f32 (25600)
                          //        | olds [96][LDO] f32 (26112)
#define SMEM_SZ  105984

// ---------------- WMMA fragment loaders (CDNA5 wave32 layouts) ----------------
__device__ __forceinline__ v16h load_a_frag(const f16* p0, int m0, int k0, int ld) {
  const int lane = threadIdx.x & 31;
  const f16* p = p0 + (m0 + (lane & 15)) * ld + k0 + ((lane & 16) ? 8 : 0);
  v8h lo = *(const v8h*)(p);
  v8h hi = *(const v8h*)(p + 16);
  v16h a;
#pragma unroll
  for (int i = 0; i < 8; ++i) { a[i] = lo[i]; a[i + 8] = hi[i]; }
  return a;
}

__device__ __forceinline__ v16h load_b_frag(const f16* p0, int k0, int n0, int ld) {
  const int lane = threadIdx.x & 31;
  const f16* p = p0 + (n0 + (lane & 15)) * ld + k0 + ((lane & 16) ? 8 : 0);
  v8h lo = *(const v8h*)(p);
  v8h hi = *(const v8h*)(p + 16);
  v16h b;
#pragma unroll
  for (int i = 0; i < 8; ++i) { b[i] = lo[i]; b[i + 8] = hi[i]; }
  return b;
}

__device__ __forceinline__ v8f load_c_frag(const float* p0, int m0, int n0, int ld) {
  const int lane = threadIdx.x & 31;
  const int col = n0 + (lane & 15);
  const int rb  = m0 + ((lane & 16) ? 8 : 0);
  v8f c;
#pragma unroll
  for (int r = 0; r < 8; ++r) c[r] = p0[(rb + r) * ld + col];
  return c;
}

#define WMMA_F16(a, b, c) \
  __builtin_amdgcn_wmma_f32_16x16x32_f16(false, (a), false, (b), (short)0, (c), false, false)

// ---------------- weight f32 -> f16 pre-conversion into workspace ----------------
__global__ void convert_weights(const float* __restrict__ qkv_w,
                                const float* __restrict__ out_w,
                                f16* __restrict__ wsh) {
  int i = blockIdx.x * blockDim.x + threadIdx.x;
  if (i < 288 * DIM) wsh[i] = (f16)qkv_w[i];
  else if (i < 288 * DIM + DIM * DIM) wsh[i] = (f16)out_w[i - 288 * DIM];
}

// ---------------- one workgroup per 8x8 window ----------------
__global__ __launch_bounds__(256) void cswin_window_kernel(
    const float* __restrict__ x, const float* __restrict__ qkv_b,
    const float* __restrict__ conv_w, const float* __restrict__ conv_b,
    const float* __restrict__ out_b, const f16* __restrict__ wsh,
    float* __restrict__ out) {
  extern __shared__ char smem[];
  f16*   xs   = (f16*)(smem + OFF_XS);
  f16*   q    = (f16*)(smem + OFF_Q);
  f16*   kk   = (f16*)(smem + OFF_K);
  f16*   vT   = (f16*)(smem + OFF_VT);
  f16*   att  = (f16*)(smem + OFF_ATT);
  float* sc   = (float*)(smem + OFF_SC);
  float* uni  = (float*)(smem + OFF_U);   // stage / lep / olds (time-multiplexed)

  const f16* qkvw = wsh;             // [288][96] row-major == [N][K] of Wqkv^T
  const f16* outw = wsh + 288 * DIM; // [96][96]

  const int t     = threadIdx.x;
  const int lane  = t & 31;
  const int wave  = t >> 5;
  const int lcol  = lane & 15;
  const int hioff = (lane & 16) ? 8 : 0;

  const int blk = blockIdx.x;          // 8192 windows
  const int b   = blk >> 10;           // 1024 windows per image
  const int rem = blk & 1023;
  const int wh  = rem >> 5;
  const int wwi = rem & 31;
  const size_t xbase = (size_t)b * DIM * 65536 + (size_t)(wh * 8) * 256 + (size_t)(wwi * 8);

  // ---- stage input window into LDS with async global->LDS b128 copies ----
  // stage layout: [c][m] f32 (stride 64, 256B rows). 1536 chunks of 16B.
  for (int ci = t; ci < DIM * 16; ci += 256) {
    int c = ci >> 4, rh = ci & 15;
    int ty = rh >> 1, half = (rh & 1) * 4;
    unsigned lds_off = (unsigned)(uintptr_t)(uni + c * 64 + ty * 8 + half);
    const float* gp = x + xbase + (size_t)c * 65536 + ty * 256 + half;
    asm volatile("global_load_async_to_lds_b128 %0, %1, off"
                 :: "v"(lds_off), "v"((unsigned long long)(uintptr_t)gp)
                 : "memory");
  }
  asm volatile("s_wait_asynccnt 0x0" ::: "memory");
  __syncthreads();

  // ---- LDS transpose + f32->f16: xs[m][c] ----
  for (int i = t; i < TOK * DIM; i += 256) {
    int c = i >> 6, m = i & 63;
    xs[m * LDX + c] = (f16)uni[c * 64 + m];
  }
  __syncthreads();

  const float scale = 0.10206207261596575f; // 1/sqrt(96), folded into Q

  // ---- GEMM1: QKV = X @ Wqkv^T + b   (4 Mtiles x 18 Ntiles, K=96) ----
  for (int t5 = wave; t5 < 72; t5 += 8) {
    int mt = t5 / 18, nt = t5 % 18;
    int col = nt * 16 + lcol;            // [0,288)
    float bias = qkv_b[col];
    v8f acc;
#pragma unroll
    for (int r = 0; r < 8; ++r) acc[r] = bias;
#pragma unroll
    for (int kt = 0; kt < 3; ++kt) {
      v16h a  = load_a_frag(xs, mt * 16, kt * 32, LDX);
      v16h bf = load_b_frag(qkvw, kt * 32, nt * 16, DIM);
      acc = WMMA_F16(a, bf, acc);
    }
    int rb = mt * 16 + hioff;
    if (col < 96) {
#pragma unroll
      for (int r = 0; r < 8; ++r) q[(rb + r) * LDX + col] = (f16)(acc[r] * scale);
    } else if (col < 192) {
#pragma unroll
      for (int r = 0; r < 8; ++r) kk[(rb + r) * LDX + (col - 96)] = (f16)acc[r];
    } else {
#pragma unroll
      for (int r = 0; r < 8; ++r) vT[(col - 192) * LDV + rb + r] = (f16)acc[r];
    }
  }
  __syncthreads();

  // ---- scores = Qs @ K^T  (4x4 tiles, K=96); kk row-major is [N][K] of K^T ----
  for (int t5 = wave; t5 < 16; t5 += 8) {
    int mt = t5 >> 2, nt = t5 & 3;
    v8f acc = {};
#pragma unroll
    for (int kt = 0; kt < 3; ++kt) {
      v16h a  = load_a_frag(q, mt * 16, kt * 32, LDX);
      v16h bf = load_b_frag(kk, kt * 32, nt * 16, LDX);
      acc = WMMA_F16(a, bf, acc);
    }
    int rb = mt * 16 + hioff, col = nt * 16 + lcol;
#pragma unroll
    for (int r = 0; r < 8; ++r) sc[(rb + r) * LDSC + col] = acc[r];
  }
  __syncthreads();

  // ---- softmax: 4 lanes per row, shfl_xor reductions (wave32) ----
  {
    int row = t >> 2, part = (t & 3) * 16;
    const float* sr = sc + row * LDSC + part;
    float ev[16];
    float mx = -1e30f;
#pragma unroll
    for (int j = 0; j < 16; ++j) { ev[j] = sr[j]; mx = fmaxf(mx, ev[j]); }
    mx = fmaxf(mx, __shfl_xor(mx, 1, 32));
    mx = fmaxf(mx, __shfl_xor(mx, 2, 32));
    float s = 0.f;
#pragma unroll
    for (int j = 0; j < 16; ++j) { ev[j] = __expf(ev[j] - mx); s += ev[j]; }
    s += __shfl_xor(s, 1, 32);
    s += __shfl_xor(s, 2, 32);
    float inv = 1.0f / s;
#pragma unroll
    for (int j = 0; j < 16; ++j) att[row * LDV + part + j] = (f16)(ev[j] * inv);
  }
  __syncthreads();   // stage region (uni) fully consumed; safe to write lep there

  // ---- LePE depthwise 3x3 conv on V (vT is [ch][pix]) -> lep[pix][ch] f32 ----
  float* lep = uni;  // [64][LDL]
  for (int i = t; i < TOK * DIM; i += 256) {
    int c = i >> 6, pix = i & 63;
    int py = pix >> 3, px = pix & 7;
    const f16* vr = vT + c * LDV;
    const float* wr = conv_w + c * 9;
    float a = conv_b[c];
#pragma unroll
    for (int dy = -1; dy <= 1; ++dy) {
      int y = py + dy;
      if ((unsigned)y < 8u) {
#pragma unroll
        for (int dx = -1; dx <= 1; ++dx) {
          int xx = px + dx;
          if ((unsigned)xx < 8u)
            a += wr[(dy + 1) * 3 + (dx + 1)] * (float)vr[y * 8 + xx];
        }
      }
    }
    lep[pix * LDL + c] = a;
  }
  __syncthreads();

  // ---- O = attn @ V + lepe  (lepe enters as the WMMA C operand); O-hat -> xs ----
  for (int t5 = wave; t5 < 24; t5 += 8) {
    int mt = t5 / 6, nt = t5 % 6;
    v8f acc = load_c_frag(lep, mt * 16, nt * 16, LDL);
#pragma unroll
    for (int kt = 0; kt < 2; ++kt) {
      v16h a  = load_a_frag(att, mt * 16, kt * 32, LDV);
      v16h bf = load_b_frag(vT, kt * 32, nt * 16, LDV);   // vT is [N=ch][K=tok]
      acc = WMMA_F16(a, bf, acc);
    }
    int rb = mt * 16 + hioff, col = nt * 16 + lcol;
#pragma unroll
    for (int r = 0; r < 8; ++r) xs[(rb + r) * LDX + col] = (f16)acc[r];
  }
  __syncthreads();   // lep consumed; uni region becomes output transpose buffer

  // ---- out proj: Y = O-hat @ Wout^T + b -> olds[c][m] (LDS transpose) ----
  float* olds = uni; // [96][LDO]
  for (int t5 = wave; t5 < 24; t5 += 8) {
    int mt = t5 / 6, nt = t5 % 6;
    int col = nt * 16 + lcol;
    float bias = out_b[col];
    v8f acc;
#pragma unroll
    for (int r = 0; r < 8; ++r) acc[r] = bias;
#pragma unroll
    for (int kt = 0; kt < 3; ++kt) {
      v16h a  = load_a_frag(xs, mt * 16, kt * 32, LDX);
      v16h bf = load_b_frag(outw, kt * 32, nt * 16, DIM);
      acc = WMMA_F16(a, bf, acc);
    }
    int rb = mt * 16 + hioff;
#pragma unroll
    for (int r = 0; r < 8; ++r) olds[col * LDO + rb + r] = acc[r];
  }
  __syncthreads();

  // ---- coalesced writeback: 16B float4 chunks along W ----
  for (int ci = t; ci < DIM * 16; ci += 256) {
    int c = ci >> 4, rh = ci & 15;
    int ty = rh >> 1, half = (rh & 1) * 4;
    v4f val = *(const v4f*)(olds + c * LDO + ty * 8 + half);
    *(v4f*)(out + xbase + (size_t)c * 65536 + ty * 256 + half) = val;
  }
}

extern "C" void kernel_launch(void* const* d_in, const int* in_sizes, int n_in,
                              void* d_out, int out_size, void* d_ws, size_t ws_size,
                              hipStream_t stream) {
  (void)in_sizes; (void)n_in; (void)out_size; (void)ws_size;
  const float* x      = (const float*)d_in[0];
  const float* qkv_w  = (const float*)d_in[1];
  const float* qkv_b  = (const float*)d_in[2];
  const float* conv_w = (const float*)d_in[3];
  const float* conv_b = (const float*)d_in[4];
  const float* out_w  = (const float*)d_in[5];
  const float* out_b  = (const float*)d_in[6];
  float* out = (float*)d_out;
  f16*   wsh = (f16*)d_ws;

  convert_weights<<<(288 * DIM + DIM * DIM + 255) / 256, 256, 0, stream>>>(qkv_w, out_w, wsh);

  (void)hipFuncSetAttribute((const void*)cswin_window_kernel,
                            hipFuncAttributeMaxDynamicSharedMemorySize, SMEM_SZ);
  cswin_window_kernel<<<8192, 256, SMEM_SZ, stream>>>(x, qkv_b, conv_w, conv_b,
                                                      out_b, wsh, out);
}
